// StandardAttention_30614526886290
// MI455X (gfx1250) — matrix operand order repaired and checked by reading, not compiled, not run on hardware
//
#include <hip/hip_runtime.h>

// ---------------------------------------------------------------------------
// Problem constants (from reference)
// ---------------------------------------------------------------------------
#define B_SZ   2
#define S_LEN  2048
#define DIM_   2048
#define NH     32
#define NKV    8
#define HD     64
#define M_ROWS (B_SZ * S_LEN)        // 4096
#define KVDIM  (NKV * HD)            // 512

typedef __bf16 bf16_t;
typedef __attribute__((ext_vector_type(8)))  __bf16 v8bf;
typedef __attribute__((ext_vector_type(16))) __bf16 v16bf;
typedef __attribute__((ext_vector_type(8)))  float  v8f;

// ---------------------------------------------------------------------------
// WMMA fragment helpers (CDNA5 wave32 layouts, 16x16x32 bf16)
//
// A (16x32, row-major source, leading dim ld):
//   lanes 0-15 : row M=l, elems = K[0..7]   then K[16..23]
//   lanes 16-31: row M=l, elems = K[8..15]  then K[24..31]
// B (32x16) built from W (N x K row-major): lane group l holds column n=l,
//   lanes 0-15 -> K[0..7] then K[16..23], lanes 16-31 -> K[8..15] then K[24..31].
// C/D (16x16 f32): VGPR i, lanes 0-15 -> (M=i,     N=l)
//                          lanes16-31 -> (M=i+8,   N=l)
// ---------------------------------------------------------------------------
__device__ __forceinline__ v16bf load_a_frag(const bf16_t* __restrict__ base, int ld) {
    const int lane = threadIdx.x & 31;
    const int half = lane >> 4, l = lane & 15;
    const bf16_t* p = base + (size_t)l * ld + 8 * half;
    v8bf c0 = *(const v8bf*)p;
    v8bf c1 = *(const v8bf*)(p + 16);
    return __builtin_shufflevector(c0, c1, 0,1,2,3,4,5,6,7,8,9,10,11,12,13,14,15);
}

__device__ __forceinline__ v16bf load_b_frag(const bf16_t* __restrict__ base, int ld) {
    const int lane = threadIdx.x & 31;
    const int half = lane >> 4, l = lane & 15;
    const bf16_t* p = base + (size_t)l * ld + 8 * half;
    v8bf c0 = *(const v8bf*)p;
    v8bf c1 = *(const v8bf*)(p + 16);
    return __builtin_shufflevector(c0, c1, 0,1,2,3,4,5,6,7,8,9,10,11,12,13,14,15);
}

__device__ __forceinline__ v8f wmma_bf16(v16bf a, v16bf b, v8f c) {
    return __builtin_amdgcn_wmma_f32_16x16x32_bf16(false, a, false, b, (short)0, c,
                                                   false, false);
}

// ---------------------------------------------------------------------------
// f32 -> bf16 convert (grid-stride)
// ---------------------------------------------------------------------------
__global__ void cvt_f32_bf16_k(const float* __restrict__ in,
                               bf16_t* __restrict__ out, size_t n) {
    for (size_t i = blockIdx.x * (size_t)blockDim.x + threadIdx.x; i < n;
         i += (size_t)gridDim.x * blockDim.x)
        out[i] = (bf16_t)in[i];
}

// ---------------------------------------------------------------------------
// C[M,N] (f32) = A[M,K](bf16 row-major) @ W[N,K](bf16 row-major)^T
// block = 256 threads = 8 waves (2x4), wave tile 64x64 (16 WMMAs / K-step,
// 1.0 b128 loads per WMMA), block tile 128x256.
// ---------------------------------------------------------------------------
__global__ __launch_bounds__(256)
void gemm_xwT_bf16_k(const bf16_t* __restrict__ A, const bf16_t* __restrict__ W,
                     float* __restrict__ C, int M, int N, int K) {
    const int wave  = threadIdx.x >> 5;
    const int waveM = wave & 1, waveN = wave >> 1;
    const int r0 = blockIdx.x * 128 + waveM * 64;
    const int c0 = blockIdx.y * 256 + waveN * 64;
    const int lane = threadIdx.x & 31;
    const int half = lane >> 4, l = lane & 15;

    v8f acc[4][4] = {};
    for (int k0 = 0; k0 < K; k0 += 32) {
        // hint next K-slice into cache (global_prefetch_b8)
        if (k0 + 64 < K) {
            __builtin_prefetch(A + (size_t)(r0 + l) * K + k0 + 64, 0, 1);
            __builtin_prefetch(W + (size_t)(c0 + l) * K + k0 + 64, 0, 1);
        }
        v16bf a[4], bb[4];
#pragma unroll
        for (int i = 0; i < 4; ++i)
            a[i] = load_a_frag(A + (size_t)(r0 + 16 * i) * K + k0, K);
#pragma unroll
        for (int j = 0; j < 4; ++j)
            bb[j] = load_b_frag(W + (size_t)(c0 + 16 * j) * K + k0, K);
#pragma unroll
        for (int i = 0; i < 4; ++i)
#pragma unroll
            for (int j = 0; j < 4; ++j)
                acc[i][j] = wmma_bf16(a[i], bb[j], acc[i][j]);
    }
#pragma unroll
    for (int i = 0; i < 4; ++i)
#pragma unroll
        for (int j = 0; j < 4; ++j)
#pragma unroll
            for (int v = 0; v < 8; ++v) {
                int row = r0 + 16 * i + v + 8 * half;
                int col = c0 + 16 * j + l;
                C[(size_t)row * N + col] = acc[i][j][v];
            }
}

// ---------------------------------------------------------------------------
// RoPE + layout change + bf16 convert for Q and K.
//  qf [b*S+s][h*64+d]  -> qb [b][h][s][d]
//  kf [b*S+s][j*64+d]  -> kb [b][j][s][d]
// one thread per (row, rotation index i<32)
// ---------------------------------------------------------------------------
__global__ void rope_pack_k(const float* __restrict__ qf, const float* __restrict__ kf,
                            bf16_t* __restrict__ qb, bf16_t* __restrict__ kb) {
    const size_t t = blockIdx.x * (size_t)blockDim.x + threadIdx.x;
    const size_t total = (size_t)M_ROWS * (NH + NKV) * 32;
    if (t >= total) return;
    const int i   = (int)(t & 31);
    size_t row    = t >> 5;
    const int h   = (int)(row % (NH + NKV));
    row /= (NH + NKV);
    const int s   = (int)(row % S_LEN);
    const int b   = (int)(row / S_LEN);

    const float inv_freq = __expf(-((float)(2 * i) / (float)HD) * 9.210340371976184f);
    float sn, cs;
    __sincosf((float)s * inv_freq, &sn, &cs);

    if (h < NH) {
        const float* src = qf + ((size_t)(b * S_LEN + s) * DIM_) + h * HD;
        bf16_t* dst = qb + (((size_t)(b * NH + h) * S_LEN) + s) * HD;
        float x1 = src[i], x2 = src[i + 32];
        dst[i]      = (bf16_t)(x1 * cs - x2 * sn);
        dst[i + 32] = (bf16_t)(x1 * sn + x2 * cs);
    } else {
        const int j = h - NH;
        const float* src = kf + ((size_t)(b * S_LEN + s) * KVDIM) + j * HD;
        bf16_t* dst = kb + (((size_t)(b * NKV + j) * S_LEN) + s) * HD;
        float x1 = src[i], x2 = src[i + 32];
        dst[i]      = (bf16_t)(x1 * cs - x2 * sn);
        dst[i + 32] = (bf16_t)(x1 * sn + x2 * cs);
    }
}

// vf [b*S+s][j*64+d] -> vt [b][j][d][s]  (bf16)
__global__ void v_transpose_k(const float* __restrict__ vf, bf16_t* __restrict__ vt) {
    const size_t t = blockIdx.x * (size_t)blockDim.x + threadIdx.x;
    const size_t total = (size_t)M_ROWS * KVDIM;
    if (t >= total) return;
    const int d = (int)(t & (HD - 1));
    size_t r  = t >> 6;
    const int j = (int)(r % NKV);
    r /= NKV;
    const int s = (int)(r % S_LEN);
    const int b = (int)(r / S_LEN);
    vt[(((size_t)(b * NKV + j) * HD) + d) * S_LEN + s] =
        (bf16_t)vf[((size_t)(b * S_LEN + s) * KVDIM) + j * HD + d];
}

// ---------------------------------------------------------------------------
// Flash attention (causal, GQA). block = 8 waves, each wave = 16 query rows,
// block tile = 128 q rows for one (b, h). K/V tiles staged into LDS with
// GLOBAL_LOAD_ASYNC_TO_LDS_B128 (ASYNCcnt) and gated by s_wait_asynccnt.
// Output written bf16 to attn [b][s][h*64+d]  (== [M][DIM] for final GEMM).
// ---------------------------------------------------------------------------
#define LDSK 72   // padded leading dim for K / V^T tiles

__global__ __launch_bounds__(256)
void flash_attn_k(const bf16_t* __restrict__ Q,   // [B][NH][S][HD]
                  const bf16_t* __restrict__ Kk,  // [B][NKV][S][HD]
                  const bf16_t* __restrict__ Vt,  // [B][NKV][HD][S]
                  bf16_t* __restrict__ Oat)       // [B][S][DIM]
{
    __shared__ bf16_t Ksh[64 * LDSK];
    __shared__ bf16_t Vsh[64 * LDSK];
    __shared__ bf16_t Psh[8 * 16 * 64];

    const int qblk = blockIdx.x, h = blockIdx.y, b = blockIdx.z;
    const int kvh  = h >> 2;                       // N_REP = 4
    const int wave = threadIdx.x >> 5;
    const int lane = threadIdx.x & 31;
    const int half = lane >> 4, l = lane & 15;
    const int qbase = qblk * 128;
    const int qr0   = qbase + wave * 16;

    const bf16_t* qptr  = Q  + (((size_t)b * NH  + h)   * S_LEN + qr0) * HD;
    const bf16_t* kbase = Kk + (((size_t)b * NKV + kvh) * S_LEN) * HD;
    const bf16_t* vbase = Vt + (((size_t)b * NKV + kvh) * HD) * S_LEN;

    // per-thread slot in the cooperative 64x64 tile loads (256 == 64*4)
    const int srow = threadIdx.x >> 2;
    const int sseg = (threadIdx.x & 3) * 16;
    // LDS byte offsets (generic-pointer low 32 bits == LDS address)
    const unsigned kdst = (unsigned)(uintptr_t)&Ksh[srow * LDSK + sseg];
    const unsigned vdst = (unsigned)(uintptr_t)&Vsh[srow * LDSK + sseg];

    // Q fragments live in registers for the whole KV loop
    const v16bf qa0 = load_a_frag(qptr, HD);        // d = 0..31
    const v16bf qa1 = load_a_frag(qptr + 32, HD);   // d = 32..63

    v8f oacc[4] = {};
    float rmax[8], rsum[8];
#pragma unroll
    for (int i = 0; i < 8; ++i) { rmax[i] = -1e30f; rsum[i] = 0.0f; }

    const int nkb = ((qbase + 127) >> 6) + 1;       // causal upper bound
    for (int kb = 0; kb < nkb; ++kb) {
        const int ks0 = kb * 64;

        // async stage of K (64 keys x 64 d) and V^T (64 d x 64 keys) into LDS
        {
            const unsigned long long ka =
                (unsigned long long)(uintptr_t)(kbase + (size_t)(ks0 + srow) * HD + sseg);
            const unsigned long long va =
                (unsigned long long)(uintptr_t)(vbase + (size_t)srow * S_LEN + ks0 + sseg);
            asm volatile(
                "global_load_async_to_lds_b128 %0, %2, off\n\t"
                "global_load_async_to_lds_b128 %0, %2, off offset:16\n\t"
                "global_load_async_to_lds_b128 %1, %3, off\n\t"
                "global_load_async_to_lds_b128 %1, %3, off offset:16"
                :
                : "v"(kdst), "v"(vdst), "v"(ka), "v"(va)
                : "memory");
            asm volatile("s_wait_asynccnt 0" ::: "memory");
        }
        __syncthreads();

        // scores: 16 q rows x 64 keys  (4 column tiles, 2 WMMAs each)
        v8f s[4];
#pragma unroll
        for (int j = 0; j < 4; ++j) {
            v8f t = {};
            t = wmma_bf16(qa0, load_b_frag(&Ksh[(j * 16) * LDSK + 0],  LDSK), t);
            t = wmma_bf16(qa1, load_b_frag(&Ksh[(j * 16) * LDSK + 32], LDSK), t);
            s[j] = t;
        }
        // scale (1/sqrt(64)) + causal mask
#pragma unroll
        for (int j = 0; j < 4; ++j)
#pragma unroll
            for (int v = 0; v < 8; ++v) {
                const int qr = qr0 + v + 8 * half;
                const int kc = ks0 + j * 16 + l;
                const float val = s[j][v] * 0.125f;
                s[j][v] = (kc <= qr) ? val : -1e30f;
            }
        // online softmax (row stats per VGPR slot, reduced over 16-lane halves)
#pragma unroll
        for (int v = 0; v < 8; ++v) {
            float m = fmaxf(fmaxf(s[0][v], s[1][v]), fmaxf(s[2][v], s[3][v]));
            m = fmaxf(m, __shfl_xor(m, 1));
            m = fmaxf(m, __shfl_xor(m, 2));
            m = fmaxf(m, __shfl_xor(m, 4));
            m = fmaxf(m, __shfl_xor(m, 8));
            const float newm = fmaxf(rmax[v], m);
            const float corr = __expf(rmax[v] - newm);
            rmax[v] = newm;
            float psum = 0.0f;
#pragma unroll
            for (int j = 0; j < 4; ++j) {
                const float p = __expf(s[j][v] - newm);
                s[j][v] = p;
                psum += p;
            }
            psum += __shfl_xor(psum, 1);
            psum += __shfl_xor(psum, 2);
            psum += __shfl_xor(psum, 4);
            psum += __shfl_xor(psum, 8);
            rsum[v] = rsum[v] * corr + psum;
#pragma unroll
            for (int n = 0; n < 4; ++n) oacc[n][v] *= corr;
        }
        // P: C-layout -> row-major bf16 in per-wave LDS region
        bf16_t* pw = &Psh[wave * 16 * 64];
#pragma unroll
        for (int j = 0; j < 4; ++j)
#pragma unroll
            for (int v = 0; v < 8; ++v)
                pw[(v + 8 * half) * 64 + j * 16 + l] = (bf16_t)s[j][v];

        // PV: A = P (16 x 64 keys), B = V^T tile (d x keys)
        const v16bf pa0 = load_a_frag(pw, 64);
        const v16bf pa1 = load_a_frag(pw + 32, 64);
#pragma unroll
        for (int n = 0; n < 4; ++n) {
            oacc[n] = wmma_bf16(pa0, load_b_frag(&Vsh[(n * 16) * LDSK + 0],  LDSK), oacc[n]);
            oacc[n] = wmma_bf16(pa1, load_b_frag(&Vsh[(n * 16) * LDSK + 32], LDSK), oacc[n]);
        }
        __syncthreads();
    }

    // normalize + store bf16 to [b][s][h*64+d]
#pragma unroll
    for (int n = 0; n < 4; ++n)
#pragma unroll
        for (int v = 0; v < 8; ++v) {
            const int srw = qr0 + v + 8 * half;
            const int col = h * HD + n * 16 + l;
            const float o = oacc[n][v] / rsum[v];
            Oat[((size_t)b * S_LEN + srw) * DIM_ + col] = (bf16_t)o;
        }
}

// ---------------------------------------------------------------------------
// host launcher
// ---------------------------------------------------------------------------
extern "C" void kernel_launch(void* const* d_in, const int* in_sizes, int n_in,
                              void* d_out, int out_size, void* d_ws, size_t ws_size,
                              hipStream_t stream) {
    const float* x  = (const float*)d_in[0];
    const float* wq = (const float*)d_in[1];
    const float* wk = (const float*)d_in[2];
    const float* wv = (const float*)d_in[3];
    const float* wo = (const float*)d_in[4];
    float* out = (float*)d_out;

    // workspace layout (256B aligned)
    char* ws = (char*)d_ws;
    size_t off = 0;
    auto alloc = [&](size_t bytes) -> char* {
        char* p = ws + off;
        off = (off + bytes + 255) & ~(size_t)255;
        return p;
    };
    bf16_t* xb   = (bf16_t*)alloc((size_t)M_ROWS * DIM_ * 2);
    bf16_t* wqb  = (bf16_t*)alloc((size_t)DIM_ * DIM_ * 2);
    bf16_t* wkb  = (bf16_t*)alloc((size_t)KVDIM * DIM_ * 2);
    bf16_t* wvb  = (bf16_t*)alloc((size_t)KVDIM * DIM_ * 2);
    bf16_t* wob  = (bf16_t*)alloc((size_t)DIM_ * DIM_ * 2);
    float*  qf   = (float*) alloc((size_t)M_ROWS * DIM_ * 4);
    float*  kf   = (float*) alloc((size_t)M_ROWS * KVDIM * 4);
    float*  vf   = (float*) alloc((size_t)M_ROWS * KVDIM * 4);
    bf16_t* qbb  = (bf16_t*)alloc((size_t)M_ROWS * DIM_ * 2);
    bf16_t* kbb  = (bf16_t*)alloc((size_t)M_ROWS * KVDIM * 2);
    bf16_t* vtb  = (bf16_t*)alloc((size_t)M_ROWS * KVDIM * 2);
    bf16_t* attn = (bf16_t*)alloc((size_t)M_ROWS * DIM_ * 2);
    (void)ws_size; (void)n_in; (void)in_sizes; (void)out_size;

    // 1) converts
    cvt_f32_bf16_k<<<4096, 256, 0, stream>>>(x,  xb,  (size_t)M_ROWS * DIM_);
    cvt_f32_bf16_k<<<4096, 256, 0, stream>>>(wq, wqb, (size_t)DIM_ * DIM_);
    cvt_f32_bf16_k<<<1024, 256, 0, stream>>>(wk, wkb, (size_t)KVDIM * DIM_);
    cvt_f32_bf16_k<<<1024, 256, 0, stream>>>(wv, wvb, (size_t)KVDIM * DIM_);
    cvt_f32_bf16_k<<<4096, 256, 0, stream>>>(wo, wob, (size_t)DIM_ * DIM_);

    // 2) QKV projections
    gemm_xwT_bf16_k<<<dim3(M_ROWS / 128, DIM_ / 256),  256, 0, stream>>>(xb, wqb, qf, M_ROWS, DIM_,  DIM_);
    gemm_xwT_bf16_k<<<dim3(M_ROWS / 128, KVDIM / 256), 256, 0, stream>>>(xb, wkb, kf, M_ROWS, KVDIM, DIM_);
    gemm_xwT_bf16_k<<<dim3(M_ROWS / 128, KVDIM / 256), 256, 0, stream>>>(xb, wvb, vf, M_ROWS, KVDIM, DIM_);

    // 3) RoPE + repack (Q,K) and transpose (V)
    {
        const size_t total = (size_t)M_ROWS * (NH + NKV) * 32;
        rope_pack_k<<<(unsigned)((total + 255) / 256), 256, 0, stream>>>(qf, kf, qbb, kbb);
        const size_t vtot = (size_t)M_ROWS * KVDIM;
        v_transpose_k<<<(unsigned)((vtot + 255) / 256), 256, 0, stream>>>(vf, vtb);
    }

    // 4) causal flash attention
    flash_attn_k<<<dim3(S_LEN / 128, NH, B_SZ), 256, 0, stream>>>(qbb, kbb, vtb, attn);

    // 5) output projection -> f32 d_out
    gemm_xwT_bf16_k<<<dim3(M_ROWS / 128, DIM_ / 256), 256, 0, stream>>>(attn, wob, out, M_ROWS, DIM_, DIM_);
}
